// mha_84731114816395
// MI455X (gfx1250) — compile-verified, matrix operand order checked by reading, not run on hardware
//
#include <hip/hip_runtime.h>
#include <hip/hip_bf16.h>

typedef __attribute__((ext_vector_type(16))) _Float16 v16h;
typedef __attribute__((ext_vector_type(8)))  _Float16 v8h;
typedef __attribute__((ext_vector_type(8)))  float    v8f;

#define BB 8
#define SS 1024
#define DD 768
#define HH 12
#define DH 64
#define RR (BB*SS)   // 8192 rows per tensor

static __device__ __forceinline__ v16h cat16(v8h lo, v8h hi) {
    return __builtin_shufflevector(lo, hi, 0,1,2,3,4,5,6,7,8,9,10,11,12,13,14,15);
}
static __device__ __forceinline__ v8h cvt8(const float4 a, const float4 b) {
    v8h r;
    r[0]=(_Float16)a.x; r[1]=(_Float16)a.y; r[2]=(_Float16)a.z; r[3]=(_Float16)a.w;
    r[4]=(_Float16)b.x; r[5]=(_Float16)b.y; r[6]=(_Float16)b.z; r[7]=(_Float16)b.w;
    return r;
}
static __device__ __forceinline__ v8f wmma16(v16h a, v16h b, v8f c) {
    // D = A(16x32 f16) * B(32x16 f16) + C(16x16 f32)
    return __builtin_amdgcn_wmma_f32_16x16x32_f16(false, a, false, b, (short)0, c, false, false);
}

// ---------------------------------------------------------------------------
// Kernel 1: Y = relu(X @ W^T) for X in {q,k,v}, fp32 in -> f16 out.
//   z==0 -> Qp [B,S,H,Dh] (== row-major [R,D])
//   z==1 -> Kp [B,S,H,Dh]
//   z==2 -> Vt [B,H,Dh,S]  (transposed per head for the PV GEMM B-operand)
// Tile: 128(M) x 128(N), K-step 32, 8 waves arranged 2(M) x 4(N),
// each wave owns 64x32 = 4x2 WMMA tiles.
// ---------------------------------------------------------------------------
__global__ __launch_bounds__(256)
void proj_relu_kernel(const float* __restrict__ q, const float* __restrict__ k,
                      const float* __restrict__ v, const float* __restrict__ W,
                      _Float16* __restrict__ Qp, _Float16* __restrict__ Kp,
                      _Float16* __restrict__ Vt)
{
    const int z = blockIdx.z;
    const float* __restrict__ X = (z == 0) ? q : (z == 1) ? k : v;
    const int rowBase = blockIdx.y * 128;
    const int colBase = blockIdx.x * 128;

    // row stride 40 halfs = 80B (multiple of 16B -> aligned ds_load_b128)
    __shared__ __align__(16) _Float16 shA[128 * 40];
    __shared__ __align__(16) _Float16 shB[128 * 40];

    const int t    = threadIdx.x;
    const int lane = t & 31;
    const int wave = t >> 5;
    const int wm = wave >> 2;       // 0..1
    const int wn = wave & 3;        // 0..3
    const int lr = lane & 15;       // fragment row/col within 16
    const int g  = lane >> 4;       // half-wave -> K octet select

    v8f acc[4][2] = {};

    const int sr = t >> 1;          // staging row 0..127
    const int sc = (t & 1) * 16;    // staging col 0 or 16

    for (int kb = 0; kb < DD; kb += 32) {
        // ---- stage A tile: X[rowBase..+127][kb..+31] -> f16 LDS
        {
            const float* src = X + (size_t)(rowBase + sr) * DD + kb + sc;
            float4 f0 = ((const float4*)src)[0];
            float4 f1 = ((const float4*)src)[1];
            float4 f2 = ((const float4*)src)[2];
            float4 f3 = ((const float4*)src)[3];
            *(v8h*)&shA[sr * 40 + sc]     = cvt8(f0, f1);
            *(v8h*)&shA[sr * 40 + sc + 8] = cvt8(f2, f3);
            // prefetch next K-step of the streaming operand
            if (kb + 32 < DD) __builtin_prefetch(src + 32, 0, 0);
        }
        // ---- stage B tile: W[colBase+n][kb..+31] -> LDS as [n][k]
        {
            const float* src = W + (size_t)(colBase + sr) * DD + kb + sc;
            float4 f0 = ((const float4*)src)[0];
            float4 f1 = ((const float4*)src)[1];
            float4 f2 = ((const float4*)src)[2];
            float4 f3 = ((const float4*)src)[3];
            *(v8h*)&shB[sr * 40 + sc]     = cvt8(f0, f1);
            *(v8h*)&shB[sr * 40 + sc + 8] = cvt8(f2, f3);
        }
        __syncthreads();

        // ---- fragments: A lane=m holds K octets g*8 and 16+g*8
        v16h af[4], bf[2];
        #pragma unroll
        for (int i = 0; i < 4; ++i) {
            const _Float16* p = &shA[(wm * 64 + i * 16 + lr) * 40];
            af[i] = cat16(*(const v8h*)(p + g * 8), *(const v8h*)(p + 16 + g * 8));
        }
        #pragma unroll
        for (int j = 0; j < 2; ++j) {
            const _Float16* p = &shB[(wn * 32 + j * 16 + lr) * 40];
            bf[j] = cat16(*(const v8h*)(p + g * 8), *(const v8h*)(p + 16 + g * 8));
        }
        #pragma unroll
        for (int i = 0; i < 4; ++i)
            #pragma unroll
            for (int j = 0; j < 2; ++j)
                acc[i][j] = wmma16(af[i], bf[j], acc[i][j]);
        __syncthreads();
    }

    // ---- epilogue: relu + f16 store
    _Float16* __restrict__ Yqk = (z == 0) ? Qp : Kp;
    #pragma unroll
    for (int i = 0; i < 4; ++i) {
        #pragma unroll
        for (int j = 0; j < 2; ++j) {
            #pragma unroll
            for (int r = 0; r < 8; ++r) {
                const int mrow = rowBase + wm * 64 + i * 16 + r + 8 * g;
                const int ncol = colBase + wn * 32 + j * 16 + lr;
                float val = acc[i][j][r];
                val = val > 0.0f ? val : 0.0f;
                const _Float16 hv = (_Float16)val;
                if (z < 2) {
                    Yqk[(size_t)mrow * DD + ncol] = hv;
                } else {
                    const int b = mrow >> 10, s = mrow & 1023;
                    const int h = ncol >> 6,  d = ncol & 63;
                    Vt[(((size_t)(b * HH + h)) * DH + d) * SS + s] = hv;
                }
            }
        }
    }
}

// ---------------------------------------------------------------------------
// Kernel 2: fused flash attention per (b, h, 128 q-rows).
// 8 waves, each owns 16 q rows. K-blocks of 32 keys; online softmax.
// Output: attnOut fp32 [B,S,D] (pre 2*o+q / LayerNorm).
// ---------------------------------------------------------------------------
__global__ __launch_bounds__(256)
void attn_kernel(const _Float16* __restrict__ Qp, const _Float16* __restrict__ Kp,
                 const _Float16* __restrict__ Vt, float* __restrict__ O)
{
    const int qb = blockIdx.x;   // 0..7  (q block of 128)
    const int h  = blockIdx.y;   // 0..11
    const int b  = blockIdx.z;   // 0..7
    const int t = threadIdx.x, lane = t & 31, wave = t >> 5;
    const int lr = lane & 15, g = lane >> 4;
    const int qBase = qb * 128;

    __shared__ __align__(16) _Float16 shK[32 * 72];     // [kc][d], stride 144B
    __shared__ __align__(16) _Float16 shV[64 * 40];     // [d][sc], stride 80B
    __shared__ __align__(16) _Float16 shP[8][16 * 40];  // per-wave P, [m][kc]

    // ---- Q fragments for this wave's 16 rows (direct global, 16B aligned)
    const int qrow = qBase + wave * 16 + lr;
    const _Float16* qptr = Qp + (size_t)(b * SS + qrow) * DD + h * DH;
    const v16h qA0 = cat16(*(const v8h*)(qptr + g * 8),
                           *(const v8h*)(qptr + 16 + g * 8));
    const v16h qA1 = cat16(*(const v8h*)(qptr + 32 + g * 8),
                           *(const v8h*)(qptr + 48 + g * 8));

    v8f cO[4] = {};
    float m_run[8], l_lane[8];
    #pragma unroll
    for (int r = 0; r < 8; ++r) { m_run[r] = -1e30f; l_lane[r] = 0.0f; }

    const float scale = 0.125f;   // 1/sqrt(64)

    for (int kb = 0; kb < SS; kb += 32) {
        // ---- stage K block: 32 keys x 64 d
        {
            const int kc = t >> 3, dc = (t & 7) * 8;
            const _Float16* src = Kp + (size_t)(b * SS + kb + kc) * DD + h * DH + dc;
            *(v8h*)&shK[kc * 72 + dc] = *(const v8h*)src;
        }
        // ---- stage V block (already transposed): 64 d x 32 keys
        {
            const int d = t >> 2, vc = (t & 3) * 8;
            const _Float16* src = Vt + (((size_t)(b * HH + h)) * DH + d) * SS + kb + vc;
            *(v8h*)&shV[d * 40 + vc] = *(const v8h*)src;
        }
        __syncthreads();

        // ---- scores: two 16x16 tiles, K-dim = Dh = 64 (2 WMMAs each)
        v8f s0 = {}, s1 = {};
        #pragma unroll
        for (int j = 0; j < 2; ++j) {
            const _Float16* kp = &shK[(j * 16 + lr) * 72];
            v16h bK0 = cat16(*(const v8h*)(kp + g * 8),      *(const v8h*)(kp + 16 + g * 8));
            v16h bK1 = cat16(*(const v8h*)(kp + 32 + g * 8), *(const v8h*)(kp + 48 + g * 8));
            v8f sc = {};
            sc = wmma16(qA0, bK0, sc);
            sc = wmma16(qA1, bK1, sc);
            if (j == 0) s0 = sc; else s1 = sc;
        }

        // ---- online softmax per row (row = 16 lanes of a half-wave)
        #pragma unroll
        for (int r = 0; r < 8; ++r) {
            float a0 = s0[r] * scale, a1 = s1[r] * scale;
            float mx = fmaxf(a0, a1);
            mx = fmaxf(mx, __shfl_xor(mx, 1, 16));
            mx = fmaxf(mx, __shfl_xor(mx, 2, 16));
            mx = fmaxf(mx, __shfl_xor(mx, 4, 16));
            mx = fmaxf(mx, __shfl_xor(mx, 8, 16));
            const float mnew  = fmaxf(m_run[r], mx);
            const float alpha = __expf(m_run[r] - mnew);
            m_run[r] = mnew;
            const float p0 = __expf(a0 - mnew);
            const float p1 = __expf(a1 - mnew);
            l_lane[r] = l_lane[r] * alpha + p0 + p1;
            #pragma unroll
            for (int tt = 0; tt < 4; ++tt) cO[tt][r] *= alpha;
            // C-layout -> A-layout via per-wave LDS (in-order DS, wave-private)
            shP[wave][(r + 8 * g) * 40 + lr]      = (_Float16)p0;
            shP[wave][(r + 8 * g) * 40 + 16 + lr] = (_Float16)p1;
        }

        // ---- P @ V: reload P as A fragment, 4 n-tiles over Dh
        const _Float16* pp = &shP[wave][lr * 40];
        const v16h pA = cat16(*(const v8h*)(pp + g * 8), *(const v8h*)(pp + 16 + g * 8));
        #pragma unroll
        for (int tt = 0; tt < 4; ++tt) {
            const _Float16* vp = &shV[(tt * 16 + lr) * 40];
            v16h bV = cat16(*(const v8h*)(vp + g * 8), *(const v8h*)(vp + 16 + g * 8));
            cO[tt] = wmma16(pA, bV, cO[tt]);
        }
        __syncthreads();
    }

    // ---- finalize: divide by row sum, write fp32
    #pragma unroll
    for (int r = 0; r < 8; ++r) {
        float l = l_lane[r];
        l += __shfl_xor(l, 1, 16);
        l += __shfl_xor(l, 2, 16);
        l += __shfl_xor(l, 4, 16);
        l += __shfl_xor(l, 8, 16);
        const float inv = 1.0f / l;
        const int row = qBase + wave * 16 + r + 8 * g;
        #pragma unroll
        for (int tt = 0; tt < 4; ++tt) {
            const int d = tt * 16 + lr;
            O[(size_t)(b * SS + row) * DD + h * DH + d] = cO[tt][r] * inv;
        }
    }
}

// ---------------------------------------------------------------------------
// Kernel 3: o = 2*attn + q, then LayerNorm over D=768. One block per row.
// ---------------------------------------------------------------------------
__global__ __launch_bounds__(256)
void ln_kernel(const float* __restrict__ A, const float* __restrict__ q,
               const float* __restrict__ gamma, const float* __restrict__ beta,
               float* __restrict__ out)
{
    const int row = blockIdx.x;
    const int t = threadIdx.x;
    const float* a  = A + (size_t)row * DD;
    const float* qq = q + (size_t)row * DD;

    float x[3];
    float sum = 0.0f, sq = 0.0f;
    #pragma unroll
    for (int i = 0; i < 3; ++i) {
        const int c = t + i * 256;
        const float v = 2.0f * a[c] + qq[c];
        x[i] = v; sum += v; sq += v * v;
    }
    #pragma unroll
    for (int off = 16; off >= 1; off >>= 1) {
        sum += __shfl_xor(sum, off, 32);
        sq  += __shfl_xor(sq,  off, 32);
    }
    __shared__ float rs[8], rq[8];
    const int wave = t >> 5, lane = t & 31;
    if (lane == 0) { rs[wave] = sum; rq[wave] = sq; }
    __syncthreads();
    float s = 0.0f, s2 = 0.0f;
    #pragma unroll
    for (int w = 0; w < 8; ++w) { s += rs[w]; s2 += rq[w]; }

    const float mu   = s * (1.0f / (float)DD);
    const float var  = s2 * (1.0f / (float)DD) - mu * mu;
    const float rstd = rsqrtf(var + 1e-5f);
    #pragma unroll
    for (int i = 0; i < 3; ++i) {
        const int c = t + i * 256;
        out[(size_t)row * DD + c] = (x[i] - mu) * rstd * gamma[c] + beta[c];
    }
}

// ---------------------------------------------------------------------------
extern "C" void kernel_launch(void* const* d_in, const int* in_sizes, int n_in,
                              void* d_out, int out_size, void* d_ws, size_t ws_size,
                              hipStream_t stream) {
    (void)in_sizes; (void)n_in; (void)out_size; (void)ws_size;
    const float* q     = (const float*)d_in[0];
    const float* k     = (const float*)d_in[1];
    const float* v     = (const float*)d_in[2];
    const float* W     = (const float*)d_in[3];
    const float* gamma = (const float*)d_in[4];
    const float* beta  = (const float*)d_in[5];
    float* out = (float*)d_out;

    // workspace partition: Qp, Kp, Vt (f16) + attnOut (f32)
    _Float16* Qp = (_Float16*)d_ws;
    _Float16* Kp = Qp + (size_t)RR * DD;
    _Float16* Vt = Kp + (size_t)RR * DD;
    float*    Ao = (float*)(Vt + (size_t)RR * DD);

    // 1) projections: grid = (N tiles, M tiles, {q,k,v})
    proj_relu_kernel<<<dim3(DD / 128, RR / 128, 3), 256, 0, stream>>>(
        q, k, v, W, Qp, Kp, Vt);

    // 2) fused attention: grid = (q blocks, heads, batch)
    attn_kernel<<<dim3(SS / 128, HH, BB), 256, 0, stream>>>(Qp, Kp, Vt, Ao);

    // 3) residual + LayerNorm
    ln_kernel<<<dim3(RR), 256, 0, stream>>>(Ao, q, gamma, beta, out);
}
